// MoELayer_4741643895014
// MI455X (gfx1250) — matrix operand order repaired and checked by reading, not compiled, not run on hardware
//
#include <hip/hip_runtime.h>
#include <math.h>

#define TOK   4096      // B*S
#define DDIM  1024
#define HDIM  4096
#define NEXP  8
#define SLOTS 8192      // TOK * K
#define SLOTP 8320      // padded (+128) so A-tile loads never run off the buffer
#define MAXT  72        // max M tiles: 8192/128 + 8 experts

typedef __attribute__((ext_vector_type(16))) __bf16 v16bf;
typedef __attribute__((ext_vector_type(8)))  __bf16 v8bf;
typedef __attribute__((ext_vector_type(2)))  __bf16 v2bf;
typedef __attribute__((ext_vector_type(8)))  float  v8f;
typedef __attribute__((ext_vector_type(4)))  int    v4i;
typedef __attribute__((address_space(1))) v4i gv4i;   // global-AS 4xi32
typedef __attribute__((address_space(3))) v4i lv4i;   // LDS-AS 4xi32

#if defined(__has_builtin)
# if __has_builtin(__builtin_amdgcn_global_load_async_to_lds_b128)
#  define HAVE_ASYNC 1
# endif
#endif
#ifndef HAVE_ASYNC
# define HAVE_ASYNC 0
#endif

// ---------------------------------------------------------------- zero meta
__global__ void zero_meta(int* m) {
    if (threadIdx.x < 16) m[threadIdx.x] = 0;   // counts[8] + cursors[8]
}

// ---------------------------------------------------------------- router
__global__ __launch_bounds__(256) void router_kernel(
    const float* __restrict__ x, const float* __restrict__ rw,
    int* __restrict__ topk, float* __restrict__ gates,
    float* __restrict__ partials, int* __restrict__ counts)
{
    __shared__ float srw[NEXP * DDIM];     // 32 KB
    __shared__ float red[256 * NEXP];      // 8 KB, deterministic tree reduce
    int tid = threadIdx.x;
    for (int i = tid; i < NEXP * DDIM; i += 256) srw[i] = rw[i];
    __syncthreads();

    int t = blockIdx.x * 256 + tid;
    float acc[NEXP];
#pragma unroll
    for (int e = 0; e < NEXP; ++e) acc[e] = 0.f;
    const float* xp = x + (size_t)t * DDIM;
    for (int d = 0; d < DDIM; ++d) {
        float xv = xp[d];
#pragma unroll
        for (int e = 0; e < NEXP; ++e) acc[e] += xv * srw[e * DDIM + d];
    }
    float m = acc[0];
#pragma unroll
    for (int e = 1; e < NEXP; ++e) m = fmaxf(m, acc[e]);
    float p[NEXP]; float s = 0.f;
#pragma unroll
    for (int e = 0; e < NEXP; ++e) { p[e] = expf(acc[e] - m); s += p[e]; }
    float inv = 1.f / s;
#pragma unroll
    for (int e = 0; e < NEXP; ++e) { p[e] *= inv; red[tid * NEXP + e] = p[e]; }

    // top-2 (first-index-wins on ties, matching lax.top_k)
    float m1 = -1.f, m2 = -1.f; int i1 = 0, i2 = 0;
#pragma unroll
    for (int e = 0; e < NEXP; ++e) {
        float v = p[e];
        if (v > m1)      { m2 = m1; i2 = i1; m1 = v; i1 = e; }
        else if (v > m2) { m2 = v; i2 = e; }
    }
    float gs = 1.f / (m1 + m2 + 1e-8f);
    topk[t * 2]     = i1;  topk[t * 2 + 1]  = i2;
    gates[t * 2]    = m1 * gs; gates[t * 2 + 1] = m2 * gs;
    atomicAdd(&counts[i1], 1);
    atomicAdd(&counts[i2], 1);

    // deterministic reduction of probs for mean_probs
    for (int stride = 128; stride > 0; stride >>= 1) {
        __syncthreads();
        if (tid < stride) {
#pragma unroll
            for (int e = 0; e < NEXP; ++e)
                red[tid * NEXP + e] += red[(tid + stride) * NEXP + e];
        }
    }
    if (tid == 0) {
#pragma unroll
        for (int e = 0; e < NEXP; ++e) partials[blockIdx.x * NEXP + e] = red[e];
    }
}

// ---------------------------------------------------------------- scan + loss
__global__ void scan_loss_kernel(
    const int* __restrict__ counts, int* __restrict__ bases, int* __restrict__ nTiles,
    int* __restrict__ tileE, int* __restrict__ tileRow, int* __restrict__ tileRows,
    const float* __restrict__ partials, float* __restrict__ lossOut)
{
    if (threadIdx.x != 0) return;
    int base = 0, nt = 0;
    for (int e = 0; e < NEXP; ++e) {
        bases[e] = base;
        int c = counts[e];
        for (int off = 0; off < c; off += 128) {
            tileE[nt] = e; tileRow[nt] = base + off;
            int r = c - off; tileRows[nt] = r < 128 ? r : 128;
            ++nt;
        }
        base += c;
    }
    *nTiles = nt;
    float loss = 0.f;
    for (int e = 0; e < NEXP; ++e) {
        float s = 0.f;
        for (int b = 0; b < 16; ++b) s += partials[b * NEXP + e];
        float mean = s / (float)TOK;
        float frac = (float)counts[e] / (float)SLOTS;
        loss += mean * frac;
    }
    *lossOut = loss * (float)NEXP;
}

// ---------------------------------------------------------------- gather x -> bf16 slots
__global__ __launch_bounds__(128) void gather_kernel(
    const float* __restrict__ x, const int* __restrict__ topk,
    const int* __restrict__ bases, int* __restrict__ cursors,
    int* __restrict__ slotTok, __bf16* __restrict__ Xg)
{
    __shared__ int sslot[2];
    int t = blockIdx.x, tid = threadIdx.x;
    if (tid < 2) {
        int e = topk[t * 2 + tid];
        int pos = atomicAdd(&cursors[e], 1);
        int slot = bases[e] + pos;
        sslot[tid] = slot;
        slotTok[t * 2 + tid] = slot;
    }
    __syncthreads();
    int s0 = sslot[0], s1 = sslot[1];
    const float* xp = x + (size_t)t * DDIM;
    for (int d = tid; d < DDIM; d += 128) {
        __bf16 b = (__bf16)xp[d];
        Xg[(size_t)s0 * DDIM + d] = b;
        Xg[(size_t)s1 * DDIM + d] = b;
    }
}

// ---------------------------------------------------------------- expert FFN GEMM
// C[128 x 128] per block, 8 waves (wave32), each wave 2x4 WMMA 16x16x32 bf16.
// Double-buffered LDS, register-prefetch pipeline. A (bf16 rows) staged into
// LDS in WMMA fragment order via GLOBAL_LOAD_ASYNC_TO_LDS_B128 (per-lane
// global + LDS addresses give us the fragment scatter for free, tracked on
// ASYNCcnt). B (fp32 weights) converted via packed v_cvt_pk_bf16_f32 — the
// two k-rows of a pair land in one dword -> one ds_store_b32.
template<int GELU_BF16>
__global__ __launch_bounds__(256) void ffn_gemm(
    const __bf16* __restrict__ A, int lda,
    const float* __restrict__ W, const float* __restrict__ bias,
    int Kdim, int N,
    __bf16* __restrict__ outB, float* __restrict__ outF,
    const int* __restrict__ tileE, const int* __restrict__ tileRow,
    const int* __restrict__ tileRows, const int* __restrict__ nTiles)
{
    int tile = blockIdx.x;
    if (tile >= *nTiles) return;
    int e         = tileE[tile];
    int rowStart  = tileRow[tile];
    int rowsValid = tileRows[tile];
    int nStart    = blockIdx.y * 128;
    const float* Wp = W + (size_t)e * Kdim * N;
    const float* bp = bias + (size_t)e * N;

    __shared__ __align__(32) __bf16 Af[2][8][32][16];  // 16KB: [buf][16-row chunk][lane][16]
    __shared__ __align__(32) __bf16 Bf[2][8][32][16];  // 16KB: [buf][16-col chunk][lane][16]

    int tid = threadIdx.x;
    int lane = tid & 31;
    int wid  = tid >> 5;
    int waveM = (wid & 3) * 32;   // 4 waves down M
    int waveN = (wid >> 2) * 64;  // 2 waves across N

    // ---- per-thread staging geometry (fixed across k-steps)
    // A: 2 groups of 8 halfs. A layout: lane=(m%16)+16*half, half=(k>=16);
    //    element i=k%16 (matches B)
    int aRow[2], aK[2], aCh[2], aL[2], aI0[2];
#pragma unroll
    for (int g = 0; g < 2; ++g) {
        int idx = tid * 2 + g;              // 0..511
        aRow[g] = idx >> 2;                 // 0..127
        aK[g]   = (idx & 3) << 3;           // 0,8,16,24
        aCh[g]  = aRow[g] >> 4;
        aL[g]   = (aRow[g] & 15) | (aK[g] & 16);
        aI0[g]  = aK[g] & 8;                // 0 or 8
    }
    // B: 2 groups, each = k-pair (k,k+1) x 4 consecutive n.
    //    B layout: lane=(n%16)+16*(k>=16), element i=k%16 (i runs along k)
    int bK[2], bN0[2];
#pragma unroll
    for (int g = 0; g < 2; ++g) {
        int idx = tid * 2 + g;              // 0..511 = 16 k-pairs x 32 n-quads
        bK[g]  = (idx >> 5) << 1;           // 0,2,..,30
        bN0[g] = (idx & 31) << 2;           // 0..124
    }

#if !HAVE_ASYNC
    v8bf areg[2];
#endif
    float4 breg[2][2];

    auto loadStage = [&](int kb, int buf) {
#pragma unroll
        for (int g = 0; g < 2; ++g) {
            const float* bp0 = Wp + (size_t)(kb + bK[g]) * N + nStart + bN0[g];
            breg[g][0] = *(const float4*)bp0;
            breg[g][1] = *(const float4*)(bp0 + N);
        }
#pragma unroll
        for (int g = 0; g < 2; ++g) {
            const __bf16* ga = A + (size_t)(rowStart + aRow[g]) * lda + kb + aK[g];
#if HAVE_ASYNC
            __builtin_amdgcn_global_load_async_to_lds_b128(
                (gv4i*)ga,
                (lv4i*)&Af[buf][aCh[g]][aL[g]][aI0[g]],
                0, 0);
#else
            areg[g] = *(const v8bf*)ga;
#endif
        }
    };

    auto storeStage = [&](int buf) {
#if !HAVE_ASYNC
#pragma unroll
        for (int g = 0; g < 2; ++g)
            *(v8bf*)&Af[buf][aCh[g]][aL[g]][aI0[g]] = areg[g];
#endif
#pragma unroll
        for (int g = 0; g < 2; ++g) {
            int ch = bN0[g] >> 4;
            int lb = (bN0[g] & 15) | (bK[g] & 16);
            int ii = bK[g] & 15;                 // even
            float e0[4] = {breg[g][0].x, breg[g][0].y, breg[g][0].z, breg[g][0].w};
            float e1[4] = {breg[g][1].x, breg[g][1].y, breg[g][1].z, breg[g][1].w};
#pragma unroll
            for (int j = 0; j < 4; ++j) {
                v2bf pk = {(__bf16)e0[j], (__bf16)e1[j]};   // one v_cvt_pk_bf16_f32
                *(v2bf*)&Bf[buf][ch][lb + j][ii] = pk;      // one ds_store_b32
            }
        }
    };

    auto syncStage = [&]() {
#if HAVE_ASYNC
# if __has_builtin(__builtin_amdgcn_s_wait_asynccnt)
        __builtin_amdgcn_s_wait_asynccnt(0);
# else
        asm volatile("s_wait_asynccnt 0x0" ::: "memory");
# endif
#endif
        __syncthreads();
    };

    v8f zero = {};
    v8f acc[2][4];
#pragma unroll
    for (int a = 0; a < 2; ++a)
#pragma unroll
        for (int b = 0; b < 4; ++b) acc[a][b] = zero;

    const int nsteps = Kdim >> 5;
    loadStage(0, 0);
    storeStage(0);
    syncStage();

    for (int i = 0; i < nsteps; ++i) {
        int buf = i & 1;
        if (i + 1 < nsteps) loadStage((i + 1) << 5, buf ^ 1);

        int ca = waveM >> 4;
        int cb = waveN >> 4;
        v16bf a0 = *(const v16bf*)&Af[buf][ca][lane][0];
        v16bf a1 = *(const v16bf*)&Af[buf][ca + 1][lane][0];
#pragma unroll
        for (int ni = 0; ni < 4; ++ni) {
            v16bf b = *(const v16bf*)&Bf[buf][cb + ni][lane][0];
            acc[0][ni] = __builtin_amdgcn_wmma_f32_16x16x32_bf16(
                false, a0, false, b, (short)0, acc[0][ni], false, false);
            acc[1][ni] = __builtin_amdgcn_wmma_f32_16x16x32_bf16(
                false, a1, false, b, (short)0, acc[1][ni], false, false);
        }
        if (i + 1 < nsteps) {
            storeStage(buf ^ 1);
            syncStage();
        }
    }

    // ---- epilogue. C layout: VGPR r -> (M = r + 8*(lane>=16), N = lane%16)
    int mLane = (lane >> 4) << 3;
    int nLane = lane & 15;
#pragma unroll
    for (int mi = 0; mi < 2; ++mi)
#pragma unroll
        for (int ni = 0; ni < 4; ++ni)
#pragma unroll
            for (int r = 0; r < 8; ++r) {
                int m = waveM + mi * 16 + mLane + r;
                if (m < rowsValid) {
                    int col = nStart + waveN + ni * 16 + nLane;
                    float v = acc[mi][ni][r] + bp[col];
                    size_t off = (size_t)(rowStart + m) * (size_t)N + col;
                    if (GELU_BF16) {
                        float gg = 0.5f * v * (1.0f + erff(v * 0.70710678118654752f));
                        outB[off] = (__bf16)gg;
                    } else {
                        outF[off] = v;
                    }
                }
            }
}

// ---------------------------------------------------------------- combine (deterministic)
__global__ __launch_bounds__(256) void combine_kernel(
    const float* __restrict__ Ybuf, const int* __restrict__ slotTok,
    const float* __restrict__ gates, float* __restrict__ out)
{
    int t = blockIdx.x;
    int d = blockIdx.y * 256 + threadIdx.x;
    int s0 = slotTok[t * 2], s1 = slotTok[t * 2 + 1];
    float g0 = gates[t * 2], g1 = gates[t * 2 + 1];
    out[(size_t)t * DDIM + d] =
        g0 * Ybuf[(size_t)s0 * DDIM + d] + g1 * Ybuf[(size_t)s1 * DDIM + d];
}

// ---------------------------------------------------------------- launch
extern "C" void kernel_launch(void* const* d_in, const int* in_sizes, int n_in,
                              void* d_out, int out_size, void* d_ws, size_t ws_size,
                              hipStream_t stream) {
    const float* x        = (const float*)d_in[0];   // [B,S,D]
    const float* router_w = (const float*)d_in[1];   // [E,D]
    const float* w1       = (const float*)d_in[2];   // [E,D,H]
    const float* b1       = (const float*)d_in[3];   // [E,H]
    const float* w2       = (const float*)d_in[4];   // [E,H,D]
    const float* b2       = (const float*)d_in[5];   // [E,D]
    float* out = (float*)d_out;                      // [T*D] + loss

    char* p = (char*)d_ws;
    auto alloc = [&](size_t bytes) -> char* {
        char* r = p; p += (bytes + 255) & ~(size_t)255; return r;
    };
    int*    meta     = (int*)   alloc(16 * 4);             // counts[8] | cursors[8]
    int*    bases    = (int*)   alloc(NEXP * 4);
    int*    nTiles   = (int*)   alloc(4);
    int*    tileE    = (int*)   alloc(MAXT * 4);
    int*    tileRow  = (int*)   alloc(MAXT * 4);
    int*    tileRows = (int*)   alloc(MAXT * 4);
    float*  partials = (float*) alloc(16 * NEXP * 4);
    int*    topk     = (int*)   alloc((size_t)TOK * 2 * 4);
    float*  gates    = (float*) alloc((size_t)TOK * 2 * 4);
    int*    slotTok  = (int*)   alloc((size_t)TOK * 2 * 4);
    __bf16* Xg       = (__bf16*)alloc((size_t)SLOTP * DDIM * 2);   // ~17 MB
    __bf16* Hbuf     = (__bf16*)alloc((size_t)SLOTP * HDIM * 2);   // ~68 MB
    float*  Ybuf     = (float*) alloc((size_t)SLOTP * DDIM * 4);   // ~34 MB

    zero_meta<<<1, 32, 0, stream>>>(meta);
    router_kernel<<<TOK / 256, 256, 0, stream>>>(x, router_w, topk, gates,
                                                 partials, meta /*counts*/);
    scan_loss_kernel<<<1, 1, 0, stream>>>(meta, bases, nTiles, tileE, tileRow,
                                          tileRows, partials,
                                          out + (size_t)TOK * DDIM);
    gather_kernel<<<TOK, 128, 0, stream>>>(x, topk, bases, meta + 8 /*cursors*/,
                                           slotTok, Xg);
    // layer 1: Hbuf = gelu(Xg @ w1 + b1), bf16
    ffn_gemm<1><<<dim3(MAXT, HDIM / 128), 256, 0, stream>>>(
        Xg, DDIM, w1, b1, DDIM, HDIM, Hbuf, nullptr,
        tileE, tileRow, tileRows, nTiles);
    // layer 2: Ybuf = Hbuf @ w2 + b2, fp32
    ffn_gemm<0><<<dim3(MAXT, DDIM / 128), 256, 0, stream>>>(
        Hbuf, HDIM, w2, b2, HDIM, DDIM, nullptr, Ybuf,
        tileE, tileRow, tileRows, nTiles);
    combine_kernel<<<dim3(TOK, DDIM / 256), 256, 0, stream>>>(Ybuf, slotTok,
                                                              gates, out);
}
